// NodeUpdate_2860448219507
// MI455X (gfx1250) — matrix operand order repaired and checked by reading, not compiled, not run on hardware
//
#include <hip/hip_runtime.h>
#include <cmath>

// ---------------- types ----------------
typedef __attribute__((ext_vector_type(16))) __bf16 v16bf;
typedef __attribute__((ext_vector_type(8)))  __bf16 v8bf;
typedef __attribute__((ext_vector_type(4)))  __bf16 v4bf;
typedef __attribute__((ext_vector_type(8)))  float  v8f;

// ---------------- problem constants ----------------
#define BATCH   16
#define ATOMS   512
#define ATOT    (BATCH * ATOMS)   // 8192 rows
#define NBR     48
#define FN      128               // node feature dim (== edge feature dim)
#define FTOT    256               // concat dim
#define OUT2    256               // 2*FN outputs of W

// LDS layout (bytes)
#define SM_W    0                         // [128][256] bf16  = 65536
#define SM_NODE 65536                     // [16][128]  bf16  =  4096
#define SM_EDGE 69632                     // [2][48][128] bf16 = 24576
#define SM_NP   94208                     // [16][128]  f32   =  8192
#define SM_MASK 102400                    // [16][48]   f32   =  3072
#define SM_TOT  105472

__device__ __forceinline__ v4bf cvt4(float4 v) {
    v4bf o;
    o[0] = (__bf16)v.x; o[1] = (__bf16)v.y;
    o[2] = (__bf16)v.z; o[3] = (__bf16)v.w;
    return o;
}

// build a 16-element bf16 WMMA operand from two contiguous 16B LDS chunks
__device__ __forceinline__ v16bf ld2(const __bf16* p0, const __bf16* p1) {
    union { v16bf v; v8bf h[2]; } u;
    u.h[0] = *(const v8bf*)p0;
    u.h[1] = *(const v8bf*)p1;
    return u.v;
}

__device__ __forceinline__ v8f wmma_bf16(v16bf a, v16bf b, v8f c) {
    return __builtin_amdgcn_wmma_f32_16x16x32_bf16(
        /*neg_a=*/false, a, /*neg_b=*/false, b,
        /*c_mod=*/(short)0, c, /*reuse_a=*/false, /*reuse_b=*/false);
}

// =====================================================================
// Kernel 1: fused (concat GEMM + sigmoid*tanh*mask + neighbor-sum)
//   grid  = (ATOT/16)*2 blocks ; block = 256 threads (8 wave32)
//   blockIdx: chunk = bx>>1 (16 atoms), h = bx&1 (which 64-feature half)
//   output: ns[atom][feature]  (ATOT x FN, f32) in workspace
// =====================================================================
__global__ __launch_bounds__(256)
void cgcnn_gemm_gate_sum(const float* __restrict__ node,
                         const float* __restrict__ edge,
                         const float* __restrict__ mask,
                         const float* __restrict__ W,
                         const float* __restrict__ bias,
                         float* __restrict__ ns)
{
    extern __shared__ char smem[];
    __bf16* Wl  = (__bf16*)(smem + SM_W);      // [128][256] rows: 0..63 gate, 64..127 extract
    __bf16* Nb  = (__bf16*)(smem + SM_NODE);   // [16][128]
    __bf16* Eb  = (__bf16*)(smem + SM_EDGE);   // [2][48][128]
    float*  np  = (float*)(smem + SM_NP);      // [16][128] nodeproj+bias
    float*  Ml  = (float*)(smem + SM_MASK);    // [16][48]

    const int t     = threadIdx.x;
    const int lane  = t & 31;
    const int wave  = t >> 5;
    const int chunk = blockIdx.x >> 1;
    const int h     = blockIdx.x & 1;          // feature half: outputs [h*64, h*64+64)
    const int atom0 = chunk * 16;

    // ---- stage W slice (128 rows x 256 cols, f32 -> bf16) ----
    #pragma unroll
    for (int i = 0; i < 32; ++i) {
        int lin = t + i * 256;                 // 8192 float4
        int r   = lin >> 6;                    // 0..127  (local row)
        int c4  = lin & 63;                    // float4 within row
        int gr  = (r < 64) ? (h * 64 + r) : (128 + h * 64 + (r - 64));
        float4 v = *(const float4*)(W + (size_t)gr * FTOT + c4 * 4);
        *(v4bf*)(Wl + r * FTOT + c4 * 4) = cvt4(v);
    }
    // ---- stage node rows (16 x 128) ----
    #pragma unroll
    for (int i = 0; i < 2; ++i) {
        int lin = t + i * 256;                 // 512 float4
        float4 v = *(const float4*)(node + (size_t)atom0 * FN + lin * 4);
        *(v4bf*)(Nb + lin * 4) = cvt4(v);
    }
    // ---- stage mask (16 x 48) ----
    #pragma unroll
    for (int i = 0; i < 3; ++i) {
        int lin = t + i * 256;                 // 768 floats
        Ml[lin] = mask[(size_t)atom0 * NBR + lin];
    }
    __syncthreads();

    // ---- nodeproj: 16 atoms x 128 local outputs, K = 128 (node part of W) ----
    {
        const int n  = lane & 15;              // atom row / output col within tile
        const int kh = lane >> 4;
        v8f c = {0.f,0.f,0.f,0.f,0.f,0.f,0.f,0.f};
        #pragma unroll
        for (int kk = 0; kk < 4; ++kk) {
            const int k0 = kk * 32;
            v16bf a = ld2(Nb + n * FN + k0 + kh * 16,
                          Nb + n * FN + k0 + kh * 16 + 8);
            v16bf b = ld2(Wl + (wave * 16 + n) * FTOT + k0 + kh * 16,
                          Wl + (wave * 16 + n) * FTOT + k0 + kh * 16 + 8);
            c = wmma_bf16(a, b, c);
        }
        const int col = wave * 16 + n;         // 0..127 local output
        const int o   = (col < 64) ? (h * 64 + col) : (128 + h * 64 + (col - 64));
        const float bv = bias[o];
        #pragma unroll
        for (int v = 0; v < 8; ++v)
            np[(kh * 8 + v) * FN + col] = c[v] + bv;   // row = atom (m), col = output
    }
    __syncthreads();

    // ---- per-atom-pair edge GEMM + gated neighbor reduction ----
    const int aL = wave >> 2;                  // which atom of the staged pair
    const int j  = wave & 3;                   // gate N-tile (pairs with extract tile j+4)
    const int n  = lane & 15;
    const int kh = lane >> 4;

    for (int pp = 0; pp < 8; ++pp) {
        // stage 2 atoms of edges: 2*48*128 f32 -> bf16
        const float* esrc = edge + (size_t)(atom0 + 2 * pp) * NBR * FN;
        #pragma unroll
        for (int i = 0; i < 12; ++i) {
            int lin = t + i * 256;             // 3072 float4
            float4 v = *(const float4*)(esrc + (size_t)lin * 4);
            *(v4bf*)(Eb + lin * 4) = cvt4(v);
        }
        // prefetch next pair into L2 while we compute
        if (pp < 7) {
            const float* nxt = edge + (size_t)(atom0 + 2 * pp + 2) * NBR * FN;
            __builtin_prefetch(nxt + t * 48, 0, 0);
        }
        __syncthreads();

        const int   al    = 2 * pp + aL;               // atom index within chunk
        const int   atomg = atom0 + al;
        const float gb = np[al * FN + j * 16 + n];         // gate nodeproj+bias
        const float eb = np[al * FN + 64 + j * 16 + n];    // extract nodeproj+bias

        float sum = 0.f;
        #pragma unroll
        for (int mt = 0; mt < 3; ++mt) {               // 3 x 16 = 48 neighbors
            v8f cg = {0.f,0.f,0.f,0.f,0.f,0.f,0.f,0.f};
            v8f ce = {0.f,0.f,0.f,0.f,0.f,0.f,0.f,0.f};
            #pragma unroll
            for (int kk = 0; kk < 4; ++kk) {
                const int k0 = kk * 32;
                v16bf a  = ld2(Eb + (aL * NBR + mt * 16 + n) * FN + k0 + kh * 16,
                               Eb + (aL * NBR + mt * 16 + n) * FN + k0 + kh * 16 + 8);
                v16bf bg = ld2(Wl + (j * 16 + n) * FTOT + 128 + k0 + kh * 16,
                               Wl + (j * 16 + n) * FTOT + 128 + k0 + kh * 16 + 8);
                v16bf be = ld2(Wl + (64 + j * 16 + n) * FTOT + 128 + k0 + kh * 16,
                               Wl + (64 + j * 16 + n) * FTOT + 128 + k0 + kh * 16 + 8);
                cg = wmma_bf16(a, bg, cg);
                ce = wmma_bf16(a, be, ce);
            }
            #pragma unroll
            for (int v = 0; v < 8; ++v) {
                const int nb = mt * 16 + kh * 8 + v;   // neighbor index
                const float mk = Ml[al * NBR + nb];
                const float g  = 1.0f / (1.0f + __expf(-(cg[v] + gb)));
                const float e  = tanhf(ce[v] + eb);
                sum += g * e * mk;
            }
        }
        // combine the two lane halves (M = 0..7 with M = 8..15)
        sum += __shfl_xor(sum, 16, 32);
        if (lane < 16)
            ns[(size_t)atomg * FN + h * 64 + j * 16 + lane] = sum;
        __syncthreads();                       // Eb reused next iteration
    }
}

// =====================================================================
// Kernel 2: BN statistics — per-feature sum / sumsq over 8192 rows
// =====================================================================
__global__ __launch_bounds__(256)
void bn_stats(const float* __restrict__ ns,
              float* __restrict__ sums, float* __restrict__ sumsq)
{
    const int f    = threadIdx.x & 127;
    const int half = threadIdx.x >> 7;
    const int r0   = blockIdx.x * 128 + half * 64;
    float s = 0.f, s2 = 0.f;
    #pragma unroll 4
    for (int i = 0; i < 64; ++i) {
        float v = ns[(size_t)(r0 + i) * FN + f];
        s  += v;
        s2 += v * v;
    }
    atomicAdd(&sums[f], s);
    atomicAdd(&sumsq[f], s2);
}

// =====================================================================
// Kernel 3: normalize + residual + tanh
// =====================================================================
__global__ __launch_bounds__(256)
void bn_apply(const float* __restrict__ node, const float* __restrict__ ns,
              const float* __restrict__ sums, const float* __restrict__ sumsq,
              const float* __restrict__ gamma, const float* __restrict__ beta,
              float* __restrict__ out)
{
    const int idx = blockIdx.x * 256 + threadIdx.x;   // < 8192*128
    const int f   = idx & 127;
    const float inv  = 1.0f / (float)ATOT;
    const float mean = sums[f] * inv;
    const float var  = sumsq[f] * inv - mean * mean;
    const float rstd = rsqrtf(var + 1e-5f);
    const float val  = (ns[idx] - mean) * rstd * gamma[f] + beta[f];
    out[idx] = tanhf(node[idx] + val);
}

// =====================================================================
extern "C" void kernel_launch(void* const* d_in, const int* in_sizes, int n_in,
                              void* d_out, int out_size, void* d_ws, size_t ws_size,
                              hipStream_t stream) {
    const float* node  = (const float*)d_in[0];  // (16,512,128)
    const float* edge  = (const float*)d_in[1];  // (16,512,48,128)
    const float* mask  = (const float*)d_in[2];  // (16,512,48)
    const float* W     = (const float*)d_in[3];  // (256,256)
    const float* bias  = (const float*)d_in[4];  // (256,)
    const float* gamma = (const float*)d_in[5];  // (128,)
    const float* beta  = (const float*)d_in[6];  // (128,)
    float* out = (float*)d_out;

    float* ns    = (float*)d_ws;                 // 8192*128 f32 = 4 MB
    float* sums  = ns + (size_t)ATOT * FN;       // 128 f32
    float* sumsq = sums + FN;                    // 128 f32

    (void)in_sizes; (void)n_in; (void)out_size; (void)ws_size;

    // allow >48KB dynamic LDS (idempotent, capture-safe host call)
    hipFuncSetAttribute((const void*)cgcnn_gemm_gate_sum,
                        hipFuncAttributeMaxDynamicSharedMemorySize, SM_TOT);

    hipMemsetAsync(sums, 0, 2 * FN * sizeof(float), stream);

    dim3 grid1((ATOT / 16) * 2);                 // 1024 blocks
    cgcnn_gemm_gate_sum<<<grid1, 256, SM_TOT, stream>>>(node, edge, mask, W, bias, ns);

    bn_stats<<<ATOT / 128, 256, 0, stream>>>(ns, sums, sumsq);

    bn_apply<<<(ATOT * FN) / 256, 256, 0, stream>>>(node, ns, sums, sumsq, gamma, beta, out);
}
